// WorkerLSTM_89807766159613
// MI455X (gfx1250) — matrix operand order repaired and checked by reading, not compiled, not run on hardware
//
#include <hip/hip_runtime.h>

// ---------------- problem constants (match reference) ----------------
constexpr int NN   = 8192;     // nodes
constexpr int BB   = 16;       // graphs
constexpr int EE   = 65536;    // edges
constexpr int NPG  = NN / BB;  // 512 nodes per graph
constexpr int HH   = 4;        // heads
constexpr int CV   = 256;      // channels
constexpr int DV   = 8;        // node feat dim
constexpr int EDV  = 5;        // edge feat dim
constexpr int HDV  = HH * CV;  // 1024

typedef __attribute__((ext_vector_type(16))) __bf16 v16bf;
typedef __attribute__((ext_vector_type(8)))  float  v8f;

static __device__ __forceinline__ __bf16 f2bf(float f) {
  union { float f; unsigned u; } x; x.f = f;
  unsigned r = x.u + 0x7FFFu + ((x.u >> 16) & 1u);   // round-to-nearest-even
  union { unsigned short s; __bf16 b; } y;
  y.s = (unsigned short)(r >> 16);
  return y.b;
}
static __device__ __forceinline__ float sigmf(float x) { return 1.0f / (1.0f + __expf(-x)); }

// A-fragment packing index: element (row r=m%16, kk=k%32) -> (lane, i)
// lane(r,half): i<8 : k = 8*half + i ; i>=8 : k = 16 + 8*half + (i-8)
static __device__ __forceinline__ void a_map(int kk, int& half, int& i) {
  if (kk < 16) { half = kk >> 3; i = kk & 7; }
  else         { half = (kk - 16) >> 3; i = 8 + ((kk - 16) & 7); }
}

// ---------------- utility kernels ----------------
__global__ void k_zero_f32(float* p, int n) {
  int i = blockIdx.x * blockDim.x + threadIdx.x;
  if (i < n) p[i] = 0.0f;
}
__global__ void k_zero_i32(int* p, int n) {
  int i = blockIdx.x * blockDim.x + threadIdx.x;
  if (i < n) p[i] = 0;
}

// ---------------- CSR build (sort edges by dst) ----------------
__global__ void k_edge_count(const int* ei, int* cnt, int E) {
  int e = blockIdx.x * blockDim.x + threadIdx.x;
  if (e < E) atomicAdd(&cnt[ei[E + e]], 1);          // dst = edge_index[1][e]
}
__global__ void k_scan(const int* cnt, int* row_ptr, int* cursor) {
  __shared__ int part[256];
  int t = threadIdx.x;
  int base = t * (NN / 256);
  int s = 0;
  for (int i = 0; i < NN / 256; ++i) s += cnt[base + i];
  part[t] = s;
  __syncthreads();
  if (t == 0) {
    int run = 0;
    for (int i = 0; i < 256; ++i) { int v = part[i]; part[i] = run; run += v; }
    row_ptr[NN] = run;
  }
  __syncthreads();
  int run = part[t];
  for (int i = 0; i < NN / 256; ++i) {
    row_ptr[base + i] = run;
    cursor[base + i]  = run;
    run += cnt[base + i];
  }
}
__global__ void k_csr_fill(const int* ei, int* cursor, int* csr_eid, int E) {
  int e = blockIdx.x * blockDim.x + threadIdx.x;
  if (e < E) {
    int dst = ei[E + e];
    int slot = atomicAdd(&cursor[dst], 1);
    csr_eid[slot] = e;
  }
}

// ---------------- fragment-order packing ----------------
// Weights W [No,K] f32 (row-major) -> B-fragment-packed bf16:
//   block (kt, nt): 32 lanes x 16 bf16 contiguous per lane
//   element (k, n): lane = ((kk>>3)&1)*16 + (n&15), i = (kk&7) + (kk>>4)*8
__global__ void k_wpack(const float* __restrict__ W, __bf16* __restrict__ P,
                        int No, int K) {
  int gid = blockIdx.x * blockDim.x + threadIdx.x;
  if (gid >= No * K) return;
  int n = gid / K, k = gid % K;
  int kt = k >> 5, kk = k & 31;
  int nt = n >> 4;
  int lane = (((kk >> 3) & 1) << 4) | (n & 15);
  int i = (kk & 7) + ((kk >> 4) << 3);
  size_t idx = (((size_t)kt * (No >> 4) + nt) * 32 + lane) * 16 + i;
  P[idx] = f2bf(W[gid]);
}

// Activations X [M,K] f32 -> A-fragment-packed bf16
__global__ void k_apack(const float* __restrict__ X, __bf16* __restrict__ P,
                        int M, int K) {
  int gid = blockIdx.x * blockDim.x + threadIdx.x;
  if (gid >= M * K) return;
  int m = gid / K, k = gid % K;
  int mt = m >> 4, r = m & 15;
  int kt = k >> 5, kk = k & 31;
  int half, i;
  a_map(kk, half, i);
  int lane = half * 16 + r;
  size_t idx = (((size_t)mt * (K >> 5) + kt) * 32 + lane) * 16 + i;
  P[idx] = f2bf(X[gid]);
}

// combined = [h, h_next[batch]] -> A-fragment-packed bf16, K = 512
__global__ void k_combined_pack(const float* __restrict__ h,
                                const float* __restrict__ hn,
                                const int* __restrict__ batch,
                                __bf16* __restrict__ P) {
  int gid = blockIdx.x * blockDim.x + threadIdx.x;
  if (gid >= NN * 2 * CV) return;
  int n = gid >> 9, c = gid & 511;
  float v = (c < CV) ? h[(size_t)n * CV + c]
                     : hn[batch[n] * CV + (c - CV)];
  int mt = n >> 4, r = n & 15;
  int kt = c >> 5, kk = c & 31;
  int half, i;
  a_map(kk, half, i);
  int lane = half * 16 + r;
  size_t idx = (((size_t)mt * (2 * CV >> 5) + kt) * 32 + lane) * 16 + i;
  P[idx] = f2bf(v);
}

// ---------------- small dense layers (K = 8) ----------------
__global__ void k_lin_small(const float* X, const float* W, const float* b,
                            float* Y, int rows, int nout, int Kd) {
  int gid = blockIdx.x * blockDim.x + threadIdx.x;
  if (gid >= rows * nout) return;
  int r = gid / nout, j = gid % nout;
  const float* xr = X + (size_t)r * Kd;
  const float* wr = W + (size_t)j * Kd;
  float acc = b[j];
  for (int k = 0; k < Kd; ++k) acc += wr[k] * xr[k];
  Y[gid] = acc;
}

// ---------------- bf16 WMMA GEMM on fragment-packed operands --------------
// C[M,N] = A[M,K] @ W.T + bias (opt ReLU). grid (M/16, N/256), block 128.
// Each wave computes a 16x64 tile: 1 A-frag load feeds 4 WMMAs per k-step.
__global__ __launch_bounds__(128) void k_gemm_bf16(
    const __bf16* __restrict__ Ap, const __bf16* __restrict__ Bp,
    const float* __restrict__ bias, float* __restrict__ C,
    int M, int N, int K, int relu) {
  int wave = threadIdx.x >> 5;
  int lane = threadIdx.x & 31;
  int half = lane >> 4;
  int r    = lane & 15;
  int mb    = blockIdx.x * 16;
  int nbase = blockIdx.y * 256 + wave * 64;
  int ktiles = K >> 5;
  int nblk   = N >> 4;

  v8f acc[4];
#pragma unroll
  for (int u = 0; u < 4; ++u) acc[u] = (v8f){0.f,0.f,0.f,0.f,0.f,0.f,0.f,0.f};

  const __bf16* pa = Ap + (((size_t)(mb >> 4) * ktiles) * 32 + lane) * 16;
  const __bf16* pb0 = Bp + ((size_t)(nbase >> 4) * 32 + lane) * 16;
  const size_t bstep = (size_t)nblk * 32 * 16;     // per-ktile stride in bf16

  for (int kt = 0; kt < ktiles; ++kt) {
    v16bf a = *(const v16bf*)pa;
    pa += 32 * 16;
    const __bf16* pb = pb0 + (size_t)kt * bstep;
#pragma unroll
    for (int u = 0; u < 4; ++u) {
      v16bf b = *(const v16bf*)(pb + (size_t)u * 32 * 16);
      acc[u] = __builtin_amdgcn_wmma_f32_16x16x32_bf16(false, a, false, b,
                                                       (short)0, acc[u],
                                                       false, false);
    }
  }
#pragma unroll
  for (int u = 0; u < 4; ++u) {
#pragma unroll
    for (int ri = 0; ri < 8; ++ri) {
      int m = mb + 8 * half + ri;
      int n = nbase + u * 16 + r;
      float v = acc[u][ri] + (bias ? bias[n] : 0.0f);
      if (relu) v = fmaxf(v, 0.0f);
      C[(size_t)m * N + n] = v;
    }
  }
}

// ---------------- GATv2 edge kernel: one wave per dst node ----------------
__global__ __launch_bounds__(256) void k_gat(
    const float* __restrict__ xl, const float* __restrict__ xr,
    const float* __restrict__ We,  const float* __restrict__ att,
    const float* __restrict__ bias, const float* __restrict__ res,
    float* __restrict__ hout,
    const int* __restrict__ row_ptr, const int* __restrict__ csr_eid,
    const int* __restrict__ ei, const float* __restrict__ ea) {
  __shared__ float We_s[HDV * EDV];   // 20 KB
  __shared__ float att_s[HH * CV];    //  4 KB
  for (int i = threadIdx.x; i < HDV * EDV; i += 256) We_s[i] = We[i];
  for (int i = threadIdx.x; i < HH * CV; i += 256)  att_s[i] = att[i];
  __syncthreads();

  int wave = threadIdx.x >> 5;
  int lane = threadIdx.x & 31;
  int node = blockIdx.x * 8 + wave;

  float xr_r[32], acc[32];
#pragma unroll
  for (int t = 0; t < 32; ++t) {
    int hh = t >> 3, j = t & 7;
    xr_r[t] = xr[(size_t)node * HDV + hh * CV + lane + 32 * j];
    acc[t]  = 0.0f;
  }
  float m[4] = {-1e30f, -1e30f, -1e30f, -1e30f};
  float s[4] = {0.f, 0.f, 0.f, 0.f};

  int beg = row_ptr[node], end = row_ptr[node + 1];
  for (int idx = beg; idx < end; ++idx) {
    int eid  = csr_eid[idx];
    int srcn = ei[eid];                       // src = edge_index[0][e]
    if (idx + 1 < end) {                      // prefetch next source row
      int e2 = csr_eid[idx + 1];
      __builtin_prefetch(xl + (size_t)ei[e2] * HDV, 0, 0);
    }
    float ea0 = ea[eid * EDV + 0], ea1 = ea[eid * EDV + 1],
          ea2 = ea[eid * EDV + 2], ea3 = ea[eid * EDV + 3],
          ea4 = ea[eid * EDV + 4];
    const float* xlp = xl + (size_t)srcn * HDV;
    float part[4] = {0.f, 0.f, 0.f, 0.f};
    float xls[32];
#pragma unroll
    for (int t = 0; t < 32; ++t) {
      int hh = t >> 3, j = t & 7;
      int jj = hh * CV + lane + 32 * j;
      float xlv = xlp[jj];
      xls[t] = xlv;
      const float* w = &We_s[jj * EDV];
      float ef = w[0] * ea0 + w[1] * ea1 + w[2] * ea2 + w[3] * ea3 + w[4] * ea4;
      float tv = xlv + xr_r[t] + ef;
      tv = tv > 0.0f ? tv : 0.2f * tv;        // leaky_relu(0.2)
      part[hh] += tv * att_s[jj];
    }
#pragma unroll
    for (int hh = 0; hh < 4; ++hh)
      for (int off = 16; off > 0; off >>= 1)
        part[hh] += __shfl_xor(part[hh], off, 32);
#pragma unroll
    for (int hh = 0; hh < 4; ++hh) {
      float mn = fmaxf(m[hh], part[hh]);
      float sc = __expf(m[hh] - mn);
      float p  = __expf(part[hh] - mn);
      s[hh] = s[hh] * sc + p;
      m[hh] = mn;
#pragma unroll
      for (int j = 0; j < 8; ++j) {
        int t = hh * 8 + j;
        acc[t] = acc[t] * sc + p * xls[t];
      }
    }
  }
#pragma unroll
  for (int j = 0; j < 8; ++j) {
    int cc = lane + 32 * j;
    float g = 0.0f;
#pragma unroll
    for (int hh = 0; hh < 4; ++hh)
      if (s[hh] > 0.0f) g += acc[hh * 8 + j] / s[hh];
    g = 0.25f * g + bias[cc];
    float e = g > 0.0f ? g : __expf(g) - 1.0f;  // elu
    hout[(size_t)node * CV + cc] = e + res[(size_t)node * CV + cc];
  }
}

// ---------------- GraphNorm (per graph, contiguous NPG blocks) ------------
__global__ __launch_bounds__(256) void k_gnorm(float* h, const float* w,
                                               const float* b, const float* a) {
  int g = blockIdx.x, c = threadIdx.x;
  float* base = h + (size_t)g * NPG * CV;
  float sum = 0.0f;
  for (int i = 0; i < NPG; ++i) sum += base[(size_t)i * CV + c];
  float mean = sum * (1.0f / NPG);
  float am = a[c] * mean;
  float v = 0.0f;
  for (int i = 0; i < NPG; ++i) {
    float d = base[(size_t)i * CV + c] - am;
    v += d * d;
  }
  float rs = rsqrtf(v * (1.0f / NPG) + 1e-5f);
  float wc = w[c], bc = b[c];
  for (int i = 0; i < NPG; ++i) {
    float d = base[(size_t)i * CV + c] - am;
    base[(size_t)i * CV + c] = d * rs * wc + bc;
  }
}

// ---------------- current-node embedding ----------------
__global__ void k_curr(const float* x, const float* h, const int* batch,
                       float* curr) {
  int n = blockIdx.x, t = threadIdx.x;
  float flag = x[(size_t)n * DV];
  if (flag == 0.0f) return;
  atomicAdd(&curr[batch[n] * CV + t], flag * h[(size_t)n * CV + t]);
}

// ---------------- LSTM cell + value head (one block per graph) ------------
__global__ __launch_bounds__(256) void k_lstm(
    const float* curr, const float* ttg, const float* h0, const float* c0,
    const float* Wih, const float* bih, const float* Whh, const float* bhh,
    const float* crW1, const float* crb1, const float* crW2, const float* crb2,
    float* outbuf) {
  int bg = blockIdx.x, t = threadIdx.x;
  __shared__ float sin_[CV + 1];
  __shared__ float sh[CV];
  __shared__ float shn[CV];
  __shared__ float red[CV];
  sin_[t] = curr[bg * CV + t];
  sh[t]   = h0[bg * CV + t];
  if (t == 0) sin_[CV] = ttg[bg];
  __syncthreads();
  float g4[4];
#pragma unroll
  for (int gi = 0; gi < 4; ++gi) {
    int row = gi * CV + t;
    float acc = bih[row] + bhh[row];
    const float* wi = Wih + (size_t)row * (CV + 1);
    for (int k = 0; k <= CV; ++k) acc += wi[k] * sin_[k];
    const float* wh = Whh + (size_t)row * CV;
    for (int k = 0; k < CV; ++k) acc += wh[k] * sh[k];
    g4[gi] = acc;
  }
  float ig = sigmf(g4[0]), fg = sigmf(g4[1]);
  float gg = tanhf(g4[2]), og = sigmf(g4[3]);
  float cn = fg * c0[bg * CV + t] + ig * gg;
  float hn = og * tanhf(cn);
  outbuf[NN + bg * CV + t]           = hn;   // h_next at offset 8192
  outbuf[NN + BB * CV + bg * CV + t] = cn;   // c_next at offset 12288
  shn[t] = hn;
  __syncthreads();
  float hid = crb1[t];
  const float* w1 = crW1 + (size_t)t * CV;
  for (int k = 0; k < CV; ++k) hid += w1[k] * shn[k];
  hid = fmaxf(hid, 0.0f);
  red[t] = hid * crW2[t];
  __syncthreads();
  for (int off = 128; off > 0; off >>= 1) {
    if (t < off) red[t] += red[t + off];
    __syncthreads();
  }
  if (t == 0) outbuf[NN + 2 * BB * CV + bg] = red[0] + crb2[0];  // value @16384
}

// ---------------- score = hid @ sc_W2.T + b2 (wave per node) --------------
__global__ __launch_bounds__(256) void k_score(const float* hid, const float* w2,
                                               const float* b2, float* scores) {
  int wave = threadIdx.x >> 5, lane = threadIdx.x & 31;
  int n = blockIdx.x * 8 + wave;
  float a = 0.0f;
#pragma unroll
  for (int j = 0; j < 8; ++j) {
    int c = lane + 32 * j;
    a += hid[(size_t)n * CV + c] * w2[c];
  }
  for (int off = 16; off > 0; off >>= 1) a += __shfl_xor(a, off, 32);
  if (lane == 0) scores[n] = a + b2[0];
}

// ================================ launcher =================================
extern "C" void kernel_launch(void* const* d_in, const int* in_sizes, int n_in,
                              void* d_out, int out_size, void* d_ws, size_t ws_size,
                              hipStream_t stream) {
  (void)in_sizes; (void)n_in; (void)out_size; (void)ws_size;
  const float* x         = (const float*)d_in[0];
  const float* edge_attr = (const float*)d_in[1];
  const float* h_state   = (const float*)d_in[2];
  const float* c_state   = (const float*)d_in[3];
  const float* ttg       = (const float*)d_in[4];
  const float* g0_Wl     = (const float*)d_in[5];
  const float* g0_bl     = (const float*)d_in[6];
  const float* g0_Wr     = (const float*)d_in[7];
  const float* g0_br     = (const float*)d_in[8];
  const float* g0_We     = (const float*)d_in[9];
  const float* g0_att    = (const float*)d_in[10];
  const float* g0_bias   = (const float*)d_in[11];
  const float* g_Wl      = (const float*)d_in[12];
  const float* g_bl      = (const float*)d_in[13];
  const float* g_Wr      = (const float*)d_in[14];
  const float* g_br      = (const float*)d_in[15];
  const float* g_We      = (const float*)d_in[16];
  const float* g_att     = (const float*)d_in[17];
  const float* g_bias    = (const float*)d_in[18];
  const float* gn_w      = (const float*)d_in[19];
  const float* gn_b      = (const float*)d_in[20];
  const float* gn_a      = (const float*)d_in[21];
  const float* inproj_W  = (const float*)d_in[22];
  const float* inproj_b  = (const float*)d_in[23];
  const float* lstm_Wih  = (const float*)d_in[24];
  const float* lstm_bih  = (const float*)d_in[25];
  const float* lstm_Whh  = (const float*)d_in[26];
  const float* lstm_bhh  = (const float*)d_in[27];
  const float* sc_W1     = (const float*)d_in[28];
  const float* sc_b1     = (const float*)d_in[29];
  const float* sc_W2     = (const float*)d_in[30];
  const float* sc_b2     = (const float*)d_in[31];
  const float* cr_W1     = (const float*)d_in[32];
  const float* cr_b1     = (const float*)d_in[33];
  const float* cr_W2     = (const float*)d_in[34];
  const float* cr_b2     = (const float*)d_in[35];
  const int*   edge_index= (const int*)d_in[36];
  const int*   batch     = (const int*)d_in[37];
  float* out = (float*)d_out;

  // ---- workspace bump allocator ----
  char* ws = (char*)d_ws;
  size_t off = 0;
  auto alloc = [&](size_t bytes) {
    size_t o = off;
    off = (off + bytes + 255) & ~(size_t)255;
    return ws + o;
  };
  float*  xl      = (float*)alloc((size_t)NN * HDV * 4);   // 32 MB
  float*  xr      = (float*)alloc((size_t)NN * HDV * 4);   // 32 MB
  float*  h0buf   = (float*)alloc((size_t)NN * CV * 4);    //  8 MB
  float*  h1buf   = (float*)alloc((size_t)NN * CV * 4);    //  8 MB
  __bf16* hb      = (__bf16*)alloc((size_t)NN * CV * 2);   //  4 MB (A-packed)
  __bf16* wtA     = (__bf16*)alloc((size_t)HDV * CV * 2);  // B-packed
  __bf16* wtB     = (__bf16*)alloc((size_t)HDV * CV * 2);
  __bf16* scW1p   = (__bf16*)alloc((size_t)2 * CV * CV * 2);
  int*    cnt     = (int*)alloc((size_t)NN * 4);
  int*    row_ptr = (int*)alloc((size_t)(NN + 1) * 4);
  int*    cursor  = (int*)alloc((size_t)NN * 4);
  int*    csr_eid = (int*)alloc((size_t)EE * 4);
  float*  curr    = (float*)alloc((size_t)BB * CV * 4);
  // aliases for buffers whose lifetimes don't overlap with xl/xr
  __bf16* comb = (__bf16*)xl;   // [N, 512] A-packed bf16 = 8 MB
  float*  hid  = (float*)xr;    // [N, 256] f32          = 8 MB

  // ---- 1. CSR by dst ----
  k_zero_i32<<<(NN + 255) / 256, 256, 0, stream>>>(cnt, NN);
  k_edge_count<<<(EE + 255) / 256, 256, 0, stream>>>(edge_index, cnt, EE);
  k_scan<<<1, 256, 0, stream>>>(cnt, row_ptr, cursor);
  k_csr_fill<<<(EE + 255) / 256, 256, 0, stream>>>(edge_index, cursor, csr_eid, EE);

  // ---- 2. layer 0 (K = 8, plain VALU) ----
  k_lin_small<<<(NN * CV + 255) / 256, 256, 0, stream>>>(x, inproj_W, inproj_b,
                                                         h1buf, NN, CV, DV);
  k_lin_small<<<(NN * HDV + 255) / 256, 256, 0, stream>>>(x, g0_Wl, g0_bl,
                                                          xl, NN, HDV, DV);
  k_lin_small<<<(NN * HDV + 255) / 256, 256, 0, stream>>>(x, g0_Wr, g0_br,
                                                          xr, NN, HDV, DV);
  k_gat<<<NN / 8, 256, 0, stream>>>(xl, xr, g0_We, g0_att, g0_bias,
                                    h1buf, h0buf, row_ptr, csr_eid,
                                    edge_index, edge_attr);
  k_gnorm<<<BB, CV, 0, stream>>>(h0buf, gn_w, gn_b, gn_a);

  // ---- 3. layers 1,2 (WMMA bf16 GEMMs for xl/xr) ----
  float* hin = h0buf;
  float* hout = h1buf;
  for (int L = 0; L < 2; ++L) {
    k_apack<<<(NN * CV + 255) / 256, 256, 0, stream>>>(hin, hb, NN, CV);
    k_wpack<<<(HDV * CV + 255) / 256, 256, 0, stream>>>(
        g_Wl + (size_t)L * HDV * CV, wtA, HDV, CV);
    k_wpack<<<(HDV * CV + 255) / 256, 256, 0, stream>>>(
        g_Wr + (size_t)L * HDV * CV, wtB, HDV, CV);
    dim3 gg(NN / 16, HDV / 256);
    k_gemm_bf16<<<gg, 128, 0, stream>>>(hb, wtA, g_bl + L * HDV, xl,
                                        NN, HDV, CV, 0);
    k_gemm_bf16<<<gg, 128, 0, stream>>>(hb, wtB, g_br + L * HDV, xr,
                                        NN, HDV, CV, 0);
    k_gat<<<NN / 8, 256, 0, stream>>>(xl, xr,
                                      g_We + (size_t)L * HDV * EDV,
                                      g_att + (size_t)L * HH * CV,
                                      g_bias + (size_t)L * CV,
                                      hin, hout, row_ptr, csr_eid,
                                      edge_index, edge_attr);
    k_gnorm<<<BB, CV, 0, stream>>>(hout, gn_w + (L + 1) * CV,
                                   gn_b + (L + 1) * CV, gn_a + (L + 1) * CV);
    float* tmp = hin; hin = hout; hout = tmp;   // final h ends in h0buf
  }
  float* hfin = hin;

  // ---- 4. current-node embedding + LSTM + value ----
  k_zero_f32<<<(BB * CV + 255) / 256, 256, 0, stream>>>(curr, BB * CV);
  k_curr<<<NN, CV, 0, stream>>>(x, hfin, batch, curr);
  k_lstm<<<BB, CV, 0, stream>>>(curr, ttg, h_state, c_state,
                                lstm_Wih, lstm_bih, lstm_Whh, lstm_bhh,
                                cr_W1, cr_b1, cr_W2, cr_b2, out);

  // ---- 5. score head: WMMA GEMM [N,512]x[512,256] + ReLU, then dot ----
  k_wpack<<<(CV * 2 * CV + 255) / 256, 256, 0, stream>>>(sc_W1, scW1p,
                                                         CV, 2 * CV);
  k_combined_pack<<<(NN * 2 * CV + 255) / 256, 256, 0, stream>>>(
      hfin, out + NN, batch, comb);
  dim3 gs(NN / 16, CV / 256);
  k_gemm_bf16<<<gs, 128, 0, stream>>>(comb, scW1p, sc_b1, hid,
                                      NN, CV, 2 * CV, 1);
  k_score<<<NN / 8, 256, 0, stream>>>(hid, sc_W2, sc_b2, out);
}
